// SSIMLoss_80023830659329
// MI455X (gfx1250) — matrix operand order repaired and checked by reading, not compiled, not run on hardware
//
#include <hip/hip_runtime.h>

typedef __attribute__((ext_vector_type(16))) _Float16 v16h;
typedef __attribute__((ext_vector_type(8)))  _Float16 v8h;
typedef __attribute__((ext_vector_type(2)))  _Float16 h2;
typedef __attribute__((ext_vector_type(8)))  float    v8f;

#define IMG_H 512
#define IMG_W 512
#define N_IMG 96                 // 32 batch * 3 channel
#define N_TILES (N_IMG * 1024)               // 98304 tiles of 16x16
#define WAVES_PER_BLOCK 8
#define N_BLOCKS (N_TILES / WAVES_PER_BLOCK) // 12288
#define N_PIX 25165824.0f
#define COL_STRIDE 40            // halves per column (80B: 16B-aligned, conflict-free)

union U16 { v16h v; h2 p[8]; };

__device__ __forceinline__ h2 pkh(float a, float b) {
    // builtin returns __fp16 ext_vector(2); bit-identical to h2
    return __builtin_bit_cast(h2, __builtin_amdgcn_cvt_pkrtz(a, b));
}

// Branchless zero-padded load: clamp the address, cndmask the value.
// No exec-mask branching -> pure VALU, dual-issue friendly.
__device__ __forceinline__ float ldz(const float* __restrict__ p, int r, int c) {
    int rc = min(max(r, 0), IMG_H - 1);
    int cc = min(max(c, 0), IMG_W - 1);
    float v = p[rc * IMG_W + cc];
    bool in = ((unsigned)r < (unsigned)IMG_H) & ((unsigned)c < (unsigned)IMG_W);
    return in ? v : 0.0f;
}

// ---------------------------------------------------------------------------
// Init: build the two WMMA coefficient fragments ONCE (identical for every
// wave). bh = horizontal band matrix in B layout (32x16, B[k][j]=w[k-j]);
// wv = vertical band matrix in A layout (16x32, A[m][k]=w[k-m]).
// ---------------------------------------------------------------------------
__global__ void ssim_init_tabs(float* __restrict__ ws) {
    const int lane = threadIdx.x;      // 0..31
    const int half = lane >> 4;
    const int n    = lane & 15;

    float w[11];
    {
        float s = 0.0f;
        for (int t = 0; t < 11; ++t) {
            float d = (float)(t - 5);
            float e = __expf(-d * d / 4.5f);
            w[t] = e; s += e;
        }
        float inv = 1.0f / s;
        for (int t = 0; t < 11; ++t) w[t] *= inv;
    }

    v16h bh, wv;
    for (int i = 0; i < 8; ++i) {
        // B layout: vgpr i<4: K = half*8+2i ; i>=4: K = half*8+8+2i (+1) ; N = lane&15
        int k0 = half * 8 + ((i < 4) ? 2 * i : 8 + 2 * i);
        int t0 = k0 - n, t1 = k0 + 1 - n;
        bh[2 * i]     = (_Float16)((t0 >= 0 && t0 <= 10) ? w[t0] : 0.0f);
        bh[2 * i + 1] = (_Float16)((t1 >= 0 && t1 <= 10) ? w[t1] : 0.0f);
        // A layout: vgpr i<4: K = half*8+2i ; i>=4: K = half*8+8+2i ; M = lane&15
        int kA = half * 8 + ((i < 4) ? 2 * i : 8 + 2 * i);
        int s0 = kA - n, s1 = kA + 1 - n;
        wv[2 * i]     = (_Float16)((s0 >= 0 && s0 <= 10) ? w[s0] : 0.0f);
        wv[2 * i + 1] = (_Float16)((s1 >= 0 && s1 <= 10) ? w[s1] : 0.0f);
    }
    v16h* tabs = (v16h*)ws;
    tabs[lane]      = bh;
    tabs[32 + lane] = wv;
}

// ---------------------------------------------------------------------------
// Main: one 16x16 output tile per wave. Separable 11-tap Gaussian via WMMA:
//   H pass: D(f16) = A(input rows, 16x32 window) x Bh   (rows 0-15 and 16-31)
//           -> f16 D layout is 8 packed consecutive rows = direct ds_store_b128
//   V pass: D(f32) = Wv x B(LDS column-major blurred rows)
// 15 WMMAs / tile for the 5 blurred moments.
// ---------------------------------------------------------------------------
__global__ __launch_bounds__(256) void ssim_main(const float* __restrict__ X,
                                                 const float* __restrict__ Y,
                                                 const float* __restrict__ tabs,
                                                 float* __restrict__ partial) {
    __shared__ __attribute__((aligned(16))) _Float16 hbLds[WAVES_PER_BLOCK][16 * COL_STRIDE];
    __shared__ float waveSum[WAVES_PER_BLOCK];

    const int lane = threadIdx.x & 31;
    const int wave = threadIdx.x >> 5;
    const int tile = blockIdx.x * WAVES_PER_BLOCK + wave;
    const int img  = tile >> 10;
    const int tr   = (tile >> 5) & 31;
    const int tc   = tile & 31;
    const int row0 = tr * 16;
    const int col0 = tc * 16;
    const float* __restrict__ xb = X + (size_t)img * (IMG_H * IMG_W);
    const float* __restrict__ yb = Y + (size_t)img * (IMG_H * IMG_W);

    const int half = lane >> 4;
    const int lrow = lane & 15;
    const int ncol = lane & 15;

    // coefficient fragments: two global_load_b128 each, L2-resident
    const v16h bh = ((const v16h*)tabs)[lane];
    const v16h wv = ((const v16h*)tabs)[32 + lane];

    // ---- load input fragments (A layout), 2 row blocks: rows -5..10, 11..26 rel
    v16h ax[2], ay[2];
    const bool interior = (tr >= 1) & (tr <= 30) & (tc >= 1) & (tc <= 30);
    if (interior) {
        #pragma unroll
        for (int b = 0; b < 2; ++b) {
            const float* __restrict__ px =
                xb + (size_t)(row0 - 5 + b * 16 + lrow) * IMG_W + (col0 - 5 + half * 8);
            const float* __restrict__ py =
                yb + (size_t)(row0 - 5 + b * 16 + lrow) * IMG_W + (col0 - 5 + half * 8);
            U16 ux, uy;
            #pragma unroll
            for (int i = 0; i < 8; ++i) {
                int off = (i < 4) ? 2 * i : 8 + 2 * i;   // constant -> folded into IOFFSET
                ux.p[i] = pkh(px[off], px[off + 1]);
                uy.p[i] = pkh(py[off], py[off + 1]);
            }
            ax[b] = ux.v; ay[b] = uy.v;
        }
    } else {
        #pragma unroll
        for (int b = 0; b < 2; ++b) {
            int r = row0 - 5 + b * 16 + lrow;
            U16 ux, uy;
            #pragma unroll
            for (int i = 0; i < 8; ++i) {
                int c = col0 - 5 + half * 8 + ((i < 4) ? 2 * i : 8 + 2 * i);
                ux.p[i] = pkh(ldz(xb, r, c), ldz(xb, r, c + 1));
                uy.p[i] = pkh(ldz(yb, r, c), ldz(yb, r, c + 1));
            }
            ax[b] = ux.v; ay[b] = uy.v;
        }
    }

    _Float16* __restrict__ hb = &hbLds[wave][0];
    v8f res[5];

    #pragma unroll
    for (int q = 0; q < 5; ++q) {
        // ---- horizontal pass (f16 accumulate -> D is 8 packed rows) ----
        #pragma unroll
        for (int b = 0; b < 2; ++b) {
            v16h a;
            if      (q == 0) a = ax[b];
            else if (q == 1) a = ay[b];
            else if (q == 2) a = ax[b] * ax[b];   // v_pk_mul_f16
            else if (q == 3) a = ay[b] * ay[b];
            else             a = ax[b] * ay[b];
            v8h cz = {};
            // f16 C/D layout: vgpr i = rows 2i,2i+1 (lanes<16: M0..7, lanes>=16: M8..15)
            // -> exactly 8 consecutive rows of column `ncol`: one ds_store_b128
            v8h d = __builtin_amdgcn_wmma_f16_16x16x32_f16(
                false, a, false, bh, (short)0, cz, false, false);
            *(v8h*)&hb[ncol * COL_STRIDE + b * 16 + half * 8] = d;   // 16B aligned
        }
        // ---- gather V-pass B fragment: pairs (K0,K0+1) contiguous -> b32/b128 loads
        // same-wave DS ops are in-order; no barrier needed
        U16 hf;
        #pragma unroll
        for (int i = 0; i < 8; ++i) {
            int k0 = half * 8 + ((i < 4) ? 2 * i : 8 + 2 * i); // rows 0..31 all freshly written
            hf.p[i] = *(const h2*)&hb[ncol * COL_STRIDE + k0];
        }
        // ---- vertical pass (f32 accumulate for the moment differences) ----
        v8f cz = {};
        res[q] = __builtin_amdgcn_wmma_f32_16x16x32_f16(
            false, wv, false, hf.v, (short)0, cz, false, false);
    }

    // ---- elementwise SSIM map + local accumulate (8 pixels per lane) ----
    const float C1 = 0.0001f;   // (0.01*1.0)^2
    const float C2 = 0.0009f;   // (0.03*1.0)^2
    float acc = 0.0f;
    #pragma unroll
    for (int i = 0; i < 8; ++i) {
        float mx  = res[0][i], my  = res[1][i];
        float exx = res[2][i], eyy = res[3][i], exy = res[4][i];
        float mx2 = mx * mx, my2 = my * my, mxy = mx * my;
        float sx  = exx - mx2, sy = eyy - my2, sxy = exy - mxy;
        float num = (2.0f * mxy + C1) * (2.0f * sxy + C2);
        float den = (mx2 + my2 + C1) * (sx + sy + C2);
        acc += num * __builtin_amdgcn_rcpf(den + 1e-8f);   // v_rcp_f32
    }

    // wave32 shuffle reduce, then block reduce (deterministic, no atomics)
    #pragma unroll
    for (int off = 16; off > 0; off >>= 1)
        acc += __shfl_xor(acc, off, 32);
    if (lane == 0) waveSum[wave] = acc;
    __syncthreads();
    if (threadIdx.x == 0) {
        float s = 0.0f;
        #pragma unroll
        for (int i = 0; i < WAVES_PER_BLOCK; ++i) s += waveSum[i];
        partial[blockIdx.x] = s;
    }
}

__global__ __launch_bounds__(256) void ssim_finish(const float* __restrict__ partial,
                                                   float* __restrict__ out, int nb) {
    __shared__ float red[256];
    float s = 0.0f;
    for (int i = threadIdx.x; i < nb; i += 256) s += partial[i];
    red[threadIdx.x] = s;
    __syncthreads();
    for (int off = 128; off > 0; off >>= 1) {
        if ((int)threadIdx.x < off) red[threadIdx.x] += red[threadIdx.x + off];
        __syncthreads();
    }
    if (threadIdx.x == 0) out[0] = 1.0f - red[0] / N_PIX;
}

extern "C" void kernel_launch(void* const* d_in, const int* in_sizes, int n_in,
                              void* d_out, int out_size, void* d_ws, size_t ws_size,
                              hipStream_t stream) {
    const float* x = (const float*)d_in[0];
    const float* y = (const float*)d_in[1];
    float* out     = (float*)d_out;
    float* tabs    = (float*)d_ws;                        // 2 KB fragment tables
    float* partial = (float*)((char*)d_ws + 4096);        // 48 KB block partials

    ssim_init_tabs<<<1, 32, 0, stream>>>(tabs);
    ssim_main<<<N_BLOCKS, 256, 0, stream>>>(x, y, tabs, partial);
    ssim_finish<<<1, 256, 0, stream>>>(partial, out, N_BLOCKS);
}
